// FastQuantumLLMOptimized_38826504356545
// MI455X (gfx1250) — matrix operand order repaired and checked by reading, not compiled, not run on hardware
//
#include <hip/hip_runtime.h>

// ---------------------------------------------------------------------------
// FastQuantumLLM cross-head attention for MI455X (gfx1250, wave32, WMMA).
// One wave = one (b,l) position:
//   xh  = x[pos].reshape(32,128) * patterns          (f32 -> f16, staged LDS)
//   S   = xh @ xh^T / sqrt(128)                      (v_wmma_f32_16x16x32_f16)
//   A   = softmax_rows(S)                            (f32 VALU, 1 lane/row)
//   out = A @ xh                                     (v_wmma_f32_16x16x32_f16)
// ---------------------------------------------------------------------------

typedef _Float16 h4  __attribute__((ext_vector_type(4)));
typedef _Float16 h8  __attribute__((ext_vector_type(8)));
typedef _Float16 h16 __attribute__((ext_vector_type(16)));
typedef float    f8  __attribute__((ext_vector_type(8)));

#define NH   32          // heads (rows)
#define HD   128         // head_dim (cols / K of matmul 1)
#define XH_STRIDE  136   // halves; 272B row = 16B * 17 (conflict-skewing pad)
#define XHT_STRIDE 40    // halves; 80B row  = 16B * 5
#define ATT_STRIDE 40    // halves; 80B row  = 16B * 5
#define SC_STRIDE  33    // floats; 132B row (skew pad)

#define XH_BYTES   (NH * XH_STRIDE * 2)   // 8704
#define XHT_BYTES  (HD * XHT_STRIDE * 2)  // 10240
#define SC_BYTES   (NH * SC_STRIDE * 4)   // 4224
#define ATT_BYTES  (NH * ATT_STRIDE * 2)  // 2560
#define WAVE_BYTES (XH_BYTES + XHT_BYTES + SC_BYTES + ATT_BYTES)  // 25728 (16B mult)

#define WAVES_PER_BLOCK 4
#define BLOCK_THREADS   (WAVES_PER_BLOCK * 32)

static __device__ __forceinline__ h16 cat16(h8 lo, h8 hi) {
  return __builtin_shufflevector(lo, hi, 0,1,2,3,4,5,6,7,8,9,10,11,12,13,14,15);
}

__global__ __launch_bounds__(BLOCK_THREADS, 1)
void fq_xhead_attn_kernel(const float* __restrict__ x,
                          const float* __restrict__ patterns,
                          float* __restrict__ out) {
  extern __shared__ __align__(16) char smem[];

  const int lane = threadIdx.x & 31;
  const int wave = threadIdx.x >> 5;
  const long pos = (long)blockIdx.x * WAVES_PER_BLOCK + wave;

  char* wbase = smem + wave * WAVE_BYTES;
  _Float16* __restrict__ xh  = (_Float16*)(wbase);
  _Float16* __restrict__ xhT = (_Float16*)(wbase + XH_BYTES);
  float*    __restrict__ sc  = (float*)   (wbase + XH_BYTES + XHT_BYTES);
  _Float16* __restrict__ att = (_Float16*)(wbase + XH_BYTES + XHT_BYTES + SC_BYTES);

  // -------- stage xh = x * patterns into LDS (row-major + transposed) -------
  const float* __restrict__ xp = x + pos * (long)(NH * HD);
  const int d0 = lane * 4;                       // each lane: 4 consecutive d
  #pragma unroll 8
  for (int h = 0; h < NH; ++h) {
    const float4 xv = *(const float4*)(xp + h * HD + d0);
    const float4 pv = *(const float4*)(patterns + h * HD + d0);
    h4 v;
    v[0] = (_Float16)(xv.x * pv.x);
    v[1] = (_Float16)(xv.y * pv.y);
    v[2] = (_Float16)(xv.z * pv.z);
    v[3] = (_Float16)(xv.w * pv.w);
    *(h4*)(xh + h * XH_STRIDE + d0) = v;         // ds_store_b64
    xhT[(d0 + 0) * XHT_STRIDE + h] = v[0];       // transposed copy (b16 x4)
    xhT[(d0 + 1) * XHT_STRIDE + h] = v[1];
    xhT[(d0 + 2) * XHT_STRIDE + h] = v[2];
    xhT[(d0 + 3) * XHT_STRIDE + h] = v[3];
  }
  __syncthreads();

  // Fragment lane geometry (wave32 WMMA 16x16x32 f16), as loaded here (A/B K order is this code's choice):
  //  A (16x32): lane L -> row M=L&15; K = kblk..kblk+15 contiguous,
  //             kblk = (L>>4)*16.
  //  B (32x16): lane L -> col N=L&15; K = kblk..kblk+15 contiguous,
  //             kblk = (L>>4)*16.
  //  C/D:       lane L, VGPR v -> M = v + 8*(L>>4), N = L&15.
  const int mlo   = lane & 15;
  const int khalf = (lane >> 4) * 8;
  const int kblk  = (lane >> 4) * 16;
  const int mhi   = (lane >> 4) * 8;

  // ---------------- matmul 1: scores = xh @ xh^T (32x32) -------------------
  f8 acc[2][2] = {};
  #pragma unroll
  for (int ko = 0; ko < HD; ko += 32) {
    h16 a[2], b[2];
    #pragma unroll
    for (int t = 0; t < 2; ++t) {
      const _Float16* ar = xh + (16 * t + mlo) * XH_STRIDE + ko + kblk;
      a[t] = cat16(*(const h8*)ar, *(const h8*)(ar + 8));
      const _Float16* br = xh + (16 * t + mlo) * XH_STRIDE + ko + kblk;
      b[t] = cat16(*(const h8*)br, *(const h8*)(br + 8));
    }
    #pragma unroll
    for (int ti = 0; ti < 2; ++ti)
      #pragma unroll
      for (int tj = 0; tj < 2; ++tj)
        acc[ti][tj] = __builtin_amdgcn_wmma_f32_16x16x32_f16(
            false, a[ti], false, b[tj], (short)0, acc[ti][tj], false, false);
  }

  const float scale = 0.0883883476483184f;       // 1/sqrt(128)
  #pragma unroll
  for (int ti = 0; ti < 2; ++ti)
    #pragma unroll
    for (int tj = 0; tj < 2; ++tj)
      #pragma unroll
      for (int v = 0; v < 8; ++v)
        sc[(16 * ti + mhi + v) * SC_STRIDE + 16 * tj + mlo] = acc[ti][tj][v] * scale;
  __syncthreads();

  // ---------------- softmax: one lane per score row -------------------------
  {
    const int r = lane;
    float rowv[NH];
    float m = -3.0e38f;
    #pragma unroll
    for (int j = 0; j < NH; ++j) {
      rowv[j] = sc[r * SC_STRIDE + j];
      m = fmaxf(m, rowv[j]);
    }
    float s = 0.0f;
    #pragma unroll
    for (int j = 0; j < NH; ++j) {
      const float e = __expf(rowv[j] - m);
      rowv[j] = e;
      s += e;
    }
    const float inv = __frcp_rn(s);
    #pragma unroll
    for (int j = 0; j < NH; ++j)
      att[r * ATT_STRIDE + j] = (_Float16)(rowv[j] * inv);
  }
  __syncthreads();

  // ---------------- matmul 2: out = attn @ xh (32x128) ---------------------
  h16 afrag[2];
  #pragma unroll
  for (int t = 0; t < 2; ++t) {
    const _Float16* ar = att + (16 * t + mlo) * ATT_STRIDE + kblk;
    afrag[t] = cat16(*(const h8*)ar, *(const h8*)(ar + 8));
  }

  float* __restrict__ op = out + pos * (long)(NH * HD);
  #pragma unroll
  for (int nj = 0; nj < HD / 16; ++nj) {
    // B[k][n] = xh[k][16*nj+n] = xhT[16*nj+n][k] -> contiguous in xhT
    const _Float16* br = xhT + (16 * nj + mlo) * XHT_STRIDE + kblk;
    const h16 b = cat16(*(const h8*)br, *(const h8*)(br + 8));
    #pragma unroll
    for (int ti = 0; ti < 2; ++ti) {
      f8 c = {};
      c = __builtin_amdgcn_wmma_f32_16x16x32_f16(
          false, afrag[ti], false, b, (short)0, c, false, false);
      #pragma unroll
      for (int v = 0; v < 8; ++v)
        op[(16 * ti + mhi + v) * HD + 16 * nj + mlo] = c[v];
    }
  }
}

extern "C" void kernel_launch(void* const* d_in, const int* in_sizes, int n_in,
                              void* d_out, int out_size, void* d_ws, size_t ws_size,
                              hipStream_t stream) {
  (void)n_in; (void)out_size; (void)d_ws; (void)ws_size;
  const float* x        = (const float*)d_in[0];   // [B, L, 4096] f32
  const float* patterns = (const float*)d_in[1];   // [32, 128]    f32
  float* out            = (float*)d_out;           // [B, L, 4096] f32

  const int positions = in_sizes[0] / (NH * HD);   // B*L = 16384
  const dim3 grid(positions / WAVES_PER_BLOCK);
  const dim3 block(BLOCK_THREADS);
  const size_t shmem = (size_t)WAVE_BYTES * WAVES_PER_BLOCK;  // 102912 B

  fq_xhead_attn_kernel<<<grid, block, shmem, stream>>>(x, patterns, out);
}
